// MoE_2430951490105
// MI455X (gfx1250) — matrix operand order repaired and checked by reading, not compiled, not run on hardware
//
#include <hip/hip_runtime.h>

// ---------------------------------------------------------------------------
// MoE (router + top-2 + capacity + SwiGLU experts + combine) for gfx1250.
// Expert GEMMs (~258 GFLOP) run on v_wmma_f32_16x16x32_bf16 with fp32 accum.
// fp32 operands are converted to bf16 while staging through a double-buffered
// LDS tile: global loads for step k+1 are issued before computing step k, so
// HBM/L2 latency overlaps the WMMA pipe (one barrier per K-step).
// ---------------------------------------------------------------------------

#define DM   1024
#define DFFC 4096
#define NEXP 8
#define LDT  40            // padded bf16 LDS row stride (80B, 16B aligned)
#define EPSF 1.1920929e-07f

typedef __bf16 bf16;
typedef __attribute__((ext_vector_type(16))) __bf16 v16bf;
typedef __attribute__((ext_vector_type(8)))  __bf16 v8bf;
typedef __attribute__((ext_vector_type(8)))  float  v8f;

union BFrag { v16bf v; v8bf h[2]; };

__device__ __forceinline__ void ld8(float* r, const float* s) {
  float4 f0 = *(const float4*)s;
  float4 f1 = *(const float4*)(s + 4);
  r[0] = f0.x; r[1] = f0.y; r[2] = f0.z; r[3] = f0.w;
  r[4] = f1.x; r[5] = f1.y; r[6] = f1.z; r[7] = f1.w;
}
__device__ __forceinline__ void st8(bf16* d, const float* r) {
  #pragma unroll
  for (int q = 0; q < 8; ++q) d[q] = (bf16)r[q];
}

// ---------------- router: logits -> softmax -> top2 -> renorm --------------
__global__ __launch_bounds__(256) void router_topk(
    const float* __restrict__ x, const float* __restrict__ rw,
    int* __restrict__ a_expert, float* __restrict__ a_prob)
{
  __shared__ float hl[DM];
  __shared__ float part[256];
  __shared__ float lg[NEXP];
  const int t = blockIdx.x, tid = threadIdx.x;
  const float* xr = x + (size_t)t * DM;
  *(float4*)(hl + tid * 4) = *(const float4*)(xr + tid * 4);
  __syncthreads();
  const int wv = tid >> 5, lane = tid & 31;
  const float* wr = rw + wv * DM;
  float p = 0.f;
  for (int i = lane; i < DM; i += 32) p += hl[i] * wr[i];
  part[tid] = p;
  __syncthreads();
  if (tid < NEXP) {
    float s = 0.f;
    for (int j = 0; j < 32; ++j) s += part[tid * 32 + j];
    lg[tid] = s;
  }
  __syncthreads();
  if (tid == 0) {
    float m = lg[0];
    for (int e = 1; e < NEXP; ++e) m = fmaxf(m, lg[e]);
    float pr[NEXP]; float s = 0.f;
    for (int e = 0; e < NEXP; ++e) { pr[e] = __expf(lg[e] - m); s += pr[e]; }
    const float inv = 1.f / s;
    for (int e = 0; e < NEXP; ++e) pr[e] *= inv;
    int i0 = 0; float p0 = pr[0];
    for (int e = 1; e < NEXP; ++e) if (pr[e] > p0) { p0 = pr[e]; i0 = e; }
    int i1 = (i0 == 0) ? 1 : 0; float p1 = pr[i1];
    for (int e = 0; e < NEXP; ++e)
      if (e != i0 && pr[e] > p1) { p1 = pr[e]; i1 = e; }
    const float sn = fmaxf(p0 + p1, EPSF);
    a_expert[2 * t + 0] = i0; a_prob[2 * t + 0] = p0 / sn;
    a_expert[2 * t + 1] = i1; a_prob[2 * t + 1] = p1 / sn;
  }
}

// ------- rank within expert (== reference's stable double argsort) ---------
__global__ __launch_bounds__(256) void rank_kernel(
    const int* __restrict__ a_expert, const float* __restrict__ a_prob,
    int* __restrict__ a_slot, int A)
{
  __shared__ float w_s[8192];
  __shared__ unsigned char e_s[8192];
  for (int i = threadIdx.x; i < A; i += 256) {
    w_s[i] = a_prob[i];
    e_s[i] = (unsigned char)a_expert[i];
  }
  __syncthreads();
  const int a = blockIdx.x * 256 + threadIdx.x;
  if (a >= A) return;
  const float wa = w_s[a];
  const unsigned ea = e_s[a];
  int cnt = 0;
  for (int b = 0; b < A; b += 4) {
    const unsigned eb4 = *(const unsigned*)&e_s[b];
    const float4 wb = *(const float4*)&w_s[b];
    cnt += (((eb4      ) & 255u) == ea) && (wb.x > wa || (wb.x == wa && (b + 0) < a));
    cnt += (((eb4 >>  8) & 255u) == ea) && (wb.y > wa || (wb.y == wa && (b + 1) < a));
    cnt += (((eb4 >> 16) & 255u) == ea) && (wb.z > wa || (wb.z == wa && (b + 2) < a));
    cnt += (((eb4 >> 24) & 255u) == ea) && (wb.w > wa || (wb.w == wa && (b + 3) < a));
  }
  a_slot[a] = cnt;
}

__global__ void init_slots(int* __restrict__ slot_tok, int n) {
  int i = blockIdx.x * 256 + threadIdx.x;
  if (i < n) slot_tok[i] = -1;
}

__global__ void fill_slots(const int* __restrict__ a_expert,
                           const int* __restrict__ a_slot,
                           int* __restrict__ slot_tok, int A, int cap)
{
  int a = blockIdx.x * 256 + threadIdx.x;
  if (a >= A) return;
  int s = a_slot[a];
  if (s < cap) slot_tok[a_expert[a] * cap + s] = a >> 1;
}

__global__ void renorm_kernel(const float* __restrict__ a_prob,
                              const int* __restrict__ a_slot,
                              float* __restrict__ a_wfin, int T, int cap)
{
  int t = blockIdx.x * 256 + threadIdx.x;
  if (t >= T) return;
  float w0 = (a_slot[2 * t + 0] < cap) ? a_prob[2 * t + 0] : 0.f;
  float w1 = (a_slot[2 * t + 1] < cap) ? a_prob[2 * t + 1] : 0.f;
  float s = fmaxf(w0 + w1, EPSF);
  a_wfin[2 * t + 0] = w0 / s;
  a_wfin[2 * t + 1] = w1 / s;
}

// --------- GEMM1: h2[e,m,n] = pre * silu(gate), bf16 WMMA, fused -----------
__global__ __launch_bounds__(256) void gemm1_swiglu(
    const float* __restrict__ x, const int* __restrict__ slot_tok,
    const float* __restrict__ wpre, const float* __restrict__ wgate,
    bf16* __restrict__ h2, int cap)
{
  __shared__ bf16 As[2][128 * LDT];
  __shared__ bf16 Bp[2][64 * LDT];
  __shared__ bf16 Bg[2][64 * LDT];
  __shared__ int  tok[128];

  const int e = blockIdx.z;
  const int m0 = blockIdx.y * 128;
  const int n0 = blockIdx.x * 64;
  const int tid = threadIdx.x;
  const int lane = tid & 31, wv = tid >> 5;
  const int wm = wv & 3, wn = wv >> 2;
  const int lh = lane >> 4, lrow = lane & 15;

  if (tid < 128) {
    int m = m0 + tid;
    tok[tid] = (m < cap) ? slot_tok[e * cap + m] : -1;
  }
  __syncthreads();

  // loop-invariant staging coordinates / source pointers (gather hoisted)
  const int rA0 = tid >> 2,         cA0 = (tid & 3) << 3;
  const int rA1 = (tid + 256) >> 2, cA1 = (tid & 3) << 3;   // (tid+256)&3 == tid&3
  const int tk0 = tok[rA0], tk1 = tok[rA1];
  const float* aS0 = (tk0 >= 0) ? (x + (size_t)tk0 * DM + cA0) : nullptr;
  const float* aS1 = (tk1 >= 0) ? (x + (size_t)tk1 * DM + cA1) : nullptr;
  const float* pS  = wpre  + ((size_t)e * DFFC + (n0 + rA0)) * DM + cA0;
  const float* gS  = wgate + ((size_t)e * DFFC + (n0 + rA0)) * DM + cA0;

  float ra0[8], ra1[8], rp[8], rg[8];

  auto stage_load = [&](int k) {
    if (aS0) ld8(ra0, aS0 + k); else { for (int q = 0; q < 8; ++q) ra0[q] = 0.f; }
    if (aS1) ld8(ra1, aS1 + k); else { for (int q = 0; q < 8; ++q) ra1[q] = 0.f; }
    ld8(rp, pS + k);
    ld8(rg, gS + k);
  };
  auto stage_store = [&](int buf) {
    st8(&As[buf][rA0 * LDT + cA0], ra0);
    st8(&As[buf][rA1 * LDT + cA1], ra1);
    st8(&Bp[buf][rA0 * LDT + cA0], rp);
    st8(&Bg[buf][rA0 * LDT + cA0], rg);
  };

  v8f acc[2][2][2];
  for (int s = 0; s < 2; ++s)
    for (int i = 0; i < 2; ++i)
      for (int j = 0; j < 2; ++j)
        for (int q = 0; q < 8; ++q) acc[s][i][j][q] = 0.f;

  stage_load(0);
  stage_store(0);
  __syncthreads();

  int buf = 0;
  for (int k0 = 0; k0 < DM; k0 += 32) {
    const bool next = (k0 + 32) < DM;
    if (next) stage_load(k0 + 32);         // prefetch into registers

    BFrag af[2];
    #pragma unroll
    for (int i = 0; i < 2; ++i) {
      const bf16* p = &As[buf][(wm * 32 + i * 16 + lrow) * LDT + lh * 8];
      af[i].h[0] = *(const v8bf*)p;
      af[i].h[1] = *(const v8bf*)(p + 16);
    }
    BFrag bp[2], bg[2];
    #pragma unroll
    for (int j = 0; j < 2; ++j) {
      const int row = (wn * 32 + j * 16 + lrow) * LDT + lh * 8;
      bp[j].h[0] = *(const v8bf*)&Bp[buf][row];
      bp[j].h[1] = *(const v8bf*)&Bp[buf][row + 16];
      bg[j].h[0] = *(const v8bf*)&Bg[buf][row];
      bg[j].h[1] = *(const v8bf*)&Bg[buf][row + 16];
    }
    #pragma unroll
    for (int i = 0; i < 2; ++i)
      #pragma unroll
      for (int j = 0; j < 2; ++j) {
        acc[0][i][j] = __builtin_amdgcn_wmma_f32_16x16x32_bf16(
            false, af[i].v, false, bp[j].v, (short)0, acc[0][i][j], false, false);
        acc[1][i][j] = __builtin_amdgcn_wmma_f32_16x16x32_bf16(
            false, af[i].v, false, bg[j].v, (short)0, acc[1][i][j], false, false);
      }

    if (next) stage_store(buf ^ 1);        // convert + commit next tile
    __syncthreads();
    buf ^= 1;
  }

  // epilogue: h2 = pre * silu(gate)  (C layout: M = r + 8*half, N = lane&15)
  #pragma unroll
  for (int i = 0; i < 2; ++i)
    for (int j = 0; j < 2; ++j) {
      const int n = n0 + wn * 32 + j * 16 + lrow;
      const int mb = m0 + wm * 32 + i * 16 + lh * 8;
      #pragma unroll
      for (int r = 0; r < 8; ++r) {
        int m = mb + r;
        if (m < cap) {
          float pre = acc[0][i][j][r];
          float g   = acc[1][i][j][r];
          float sig = 1.f / (1.f + __expf(-g));
          h2[((size_t)(e * cap + m)) * DFFC + n] = (bf16)(pre * g * sig);
        }
      }
    }
}

// --------- GEMM2: expert_out[e,m,n] = h2 @ Wpost^T, bf16 WMMA --------------
__global__ __launch_bounds__(256) void gemm2_post(
    const bf16* __restrict__ h2, const float* __restrict__ wpost,
    float* __restrict__ eo, int cap)
{
  __shared__ bf16 As[2][128 * LDT];
  __shared__ bf16 Bs[2][64 * LDT];

  const int e = blockIdx.z;
  const int m0 = blockIdx.y * 128;
  const int n0 = blockIdx.x * 64;
  const int tid = threadIdx.x;
  const int lane = tid & 31, wv = tid >> 5;
  const int wm = wv & 3, wn = wv >> 2;
  const int lh = lane >> 4, lrow = lane & 15;

  const int rA0 = tid >> 2,         cA0 = (tid & 3) << 3;
  const int rA1 = (tid + 256) >> 2, cA1 = (tid & 3) << 3;
  const bf16* aS0 = (m0 + rA0 < cap)
      ? (h2 + ((size_t)(e * cap + m0 + rA0)) * DFFC + cA0) : nullptr;
  const bf16* aS1 = (m0 + rA1 < cap)
      ? (h2 + ((size_t)(e * cap + m0 + rA1)) * DFFC + cA1) : nullptr;
  const float* bS = wpost + ((size_t)e * DM + (n0 + rA0)) * DFFC + cA0;

  v8bf zv;
  for (int q = 0; q < 8; ++q) zv[q] = (bf16)0.f;
  v8bf ra0, ra1;
  float rb[8];

  auto stage_load = [&](int k) {
    ra0 = aS0 ? *(const v8bf*)(aS0 + k) : zv;
    ra1 = aS1 ? *(const v8bf*)(aS1 + k) : zv;
    ld8(rb, bS + k);
  };
  auto stage_store = [&](int buf) {
    *(v8bf*)&As[buf][rA0 * LDT + cA0] = ra0;
    *(v8bf*)&As[buf][rA1 * LDT + cA1] = ra1;
    st8(&Bs[buf][rA0 * LDT + cA0], rb);
  };

  v8f acc[2][2];
  for (int i = 0; i < 2; ++i)
    for (int j = 0; j < 2; ++j)
      for (int q = 0; q < 8; ++q) acc[i][j][q] = 0.f;

  stage_load(0);
  stage_store(0);
  __syncthreads();

  int buf = 0;
  for (int k0 = 0; k0 < DFFC; k0 += 32) {
    const bool next = (k0 + 32) < DFFC;
    if (next) stage_load(k0 + 32);

    BFrag af[2], bf_[2];
    #pragma unroll
    for (int i = 0; i < 2; ++i) {
      const bf16* p = &As[buf][(wm * 32 + i * 16 + lrow) * LDT + lh * 8];
      af[i].h[0] = *(const v8bf*)p;
      af[i].h[1] = *(const v8bf*)(p + 16);
    }
    #pragma unroll
    for (int j = 0; j < 2; ++j) {
      const int row = (wn * 32 + j * 16 + lrow) * LDT + lh * 8;
      bf_[j].h[0] = *(const v8bf*)&Bs[buf][row];
      bf_[j].h[1] = *(const v8bf*)&Bs[buf][row + 16];
    }
    #pragma unroll
    for (int i = 0; i < 2; ++i)
      #pragma unroll
      for (int j = 0; j < 2; ++j)
        acc[i][j] = __builtin_amdgcn_wmma_f32_16x16x32_bf16(
            false, af[i].v, false, bf_[j].v, (short)0, acc[i][j], false, false);

    if (next) stage_store(buf ^ 1);
    __syncthreads();
    buf ^= 1;
  }

  #pragma unroll
  for (int i = 0; i < 2; ++i)
    for (int j = 0; j < 2; ++j) {
      const int n = n0 + wn * 32 + j * 16 + lrow;
      const int mb = m0 + wm * 32 + i * 16 + lh * 8;
      #pragma unroll
      for (int r = 0; r < 8; ++r) {
        int m = mb + r;
        if (m < cap) eo[((size_t)(e * cap + m)) * DM + n] = acc[i][j][r];
      }
    }
}

// ---------------- combine: gather <=2 weighted rows per token --------------
__global__ __launch_bounds__(256) void combine_kernel(
    const int* __restrict__ a_expert, const int* __restrict__ a_slot,
    const float* __restrict__ a_wfin, const float* __restrict__ eo,
    float* __restrict__ out, int cap)
{
  const int t = blockIdx.x;
  const int c = threadIdx.x * 4;
  float4 acc = make_float4(0.f, 0.f, 0.f, 0.f);
  for (int k = 0; k < 2; ++k) {
    int a = 2 * t + k;
    int s = a_slot[a];
    if (s < cap) {
      float w = a_wfin[a];
      const float4 v = *(const float4*)(eo + ((size_t)(a_expert[a] * cap + s)) * DM + c);
      acc.x += w * v.x; acc.y += w * v.y; acc.z += w * v.z; acc.w += w * v.w;
    }
  }
  *(float4*)(out + (size_t)t * DM + c) = acc;
}

// ---------------------------------------------------------------------------
extern "C" void kernel_launch(void* const* d_in, const int* in_sizes, int n_in,
                              void* d_out, int out_size, void* d_ws, size_t ws_size,
                              hipStream_t stream)
{
  (void)n_in; (void)out_size; (void)ws_size;
  const float* x     = (const float*)d_in[0];
  const float* rw    = (const float*)d_in[1];
  const float* wpre  = (const float*)d_in[2];
  const float* wgate = (const float*)d_in[3];
  const float* wpost = (const float*)d_in[4];
  float* out = (float*)d_out;

  const int T = in_sizes[0] / DM;          // 4096
  const int A = T * 2;                     // 8192
  int cap = (5 * A + 4 * NEXP - 1) / (4 * NEXP);   // ceil(1.25*A/E) = 1280
  if (cap < 1) cap = 1;

  unsigned long long p = (unsigned long long)d_ws;
  auto alloc = [&](size_t bytes) -> void* {
    p = (p + 255ull) & ~255ull;
    void* r = (void*)p;
    p += bytes;
    return r;
  };
  int*   a_expert = (int*)  alloc((size_t)A * 4);
  float* a_prob   = (float*)alloc((size_t)A * 4);
  int*   a_slot   = (int*)  alloc((size_t)A * 4);
  float* a_wfin   = (float*)alloc((size_t)A * 4);
  int*   slot_tok = (int*)  alloc((size_t)NEXP * cap * 4);
  bf16*  h2       = (bf16*) alloc((size_t)NEXP * cap * DFFC * 2);
  float* eo       = (float*)alloc((size_t)NEXP * cap * DM * 4);

  router_topk<<<T, 256, 0, stream>>>(x, rw, a_expert, a_prob);
  rank_kernel<<<(A + 255) / 256, 256, 0, stream>>>(a_expert, a_prob, a_slot, A);
  init_slots<<<(NEXP * cap + 255) / 256, 256, 0, stream>>>(slot_tok, NEXP * cap);
  fill_slots<<<(A + 255) / 256, 256, 0, stream>>>(a_expert, a_slot, slot_tok, A, cap);
  renorm_kernel<<<(T + 255) / 256, 256, 0, stream>>>(a_prob, a_slot, a_wfin, T, cap);

  dim3 g1(DFFC / 64, (cap + 127) / 128, NEXP);
  gemm1_swiglu<<<g1, 256, 0, stream>>>(x, slot_tok, wpre, wgate, h2, cap);

  dim3 g2(DM / 64, (cap + 127) / 128, NEXP);
  gemm2_post<<<g2, 256, 0, stream>>>(h2, wpost, eo, cap);

  combine_kernel<<<T, 256, 0, stream>>>(a_expert, a_slot, a_wfin, eo, out, cap);
}
